// WordLabelAttention_39264591020668
// MI455X (gfx1250) — matrix operand order repaired and checked by reading, not compiled, not run on hardware
//
#include <hip/hip_runtime.h>
#include <hip/hip_bf16.h>

typedef __attribute__((ext_vector_type(16))) _Float16 v16h;
typedef __attribute__((ext_vector_type(8)))  _Float16 v8h;
typedef __attribute__((ext_vector_type(4)))  _Float16 v4h;
typedef __attribute__((ext_vector_type(8)))  float    v8f;

__device__ __forceinline__ v8f wmma16(v16h a, v16h b, v8f c) {
  return __builtin_amdgcn_wmma_f32_16x16x32_f16(
      false, a, false, b, (short)0, c, /*reuse_a=*/false, /*reuse_b=*/false);
}
// reuse-A hint: previous matrix op is the identical opcode with the same A.
__device__ __forceinline__ v8f wmma16_ra(v16h a, v16h b, v8f c) {
  return __builtin_amdgcn_wmma_f32_16x16x32_f16(
      false, a, false, b, (short)0, c, /*reuse_a=*/true, /*reuse_b=*/false);
}

// Two aligned 16B vector loads -> one v16h operand.
__device__ __forceinline__ v16h ld2x8(const _Float16* p0, const _Float16* p1) {
  v8h lo = *(const v8h*)p0;
  v8h hi = *(const v8h*)p1;
  v16h r;
#pragma unroll
  for (int j = 0; j < 8; ++j) { r[j] = lo[j]; r[8 + j] = hi[j]; }
  return r;
}

// LDS byte offset of a __shared__ address (flat addr[31:0] == LDS offset).
__device__ __forceinline__ unsigned lds_off(const void* p) {
  return (unsigned)(size_t)p;
}

// Async DMA: LDS[ldsoff..+15] = MEM[saddr + goff .. +15]; tracked by ASYNCcnt.
__device__ __forceinline__ void async_copy_b128(unsigned ldsoff, unsigned goff,
                                                const void* saddr) {
  asm volatile("global_load_async_to_lds_b128 %0, %1, %2"
               :: "v"(ldsoff), "v"(goff), "s"(saddr)
               : "memory");
}

__device__ __forceinline__ void wait_async0() {
  asm volatile("s_wait_asynccnt 0x0" ::: "memory");
}

// ---------------------------------------------------------------------------
// Kernel 0: Wo f32 -> f16 (one pass; outproj then loads f16 directly)
// ---------------------------------------------------------------------------
__global__ __launch_bounds__(256)
void cvtw_kernel(const float* __restrict__ W, _Float16* __restrict__ W16) {
  const int i = (blockIdx.x * 256 + threadIdx.x) * 4;
  const float4 v = *(const float4*)(W + i);
  v4h o;
  o[0] = (_Float16)v.x; o[1] = (_Float16)v.y;
  o[2] = (_Float16)v.z; o[3] = (_Float16)v.w;
  *(v4h*)(W16 + i) = o;
}

// ---------------------------------------------------------------------------
// Kernel 1: per-head projection  out[r][e] = sum_d X[r][d] * W[e][d]
// X: rows x 64 f32 (head-slices are contiguous rows), W: 64x64 f32.
// remap==1: rows are (n,kl,h) flat; store to [n][h][kl][64] layout (for K/V).
// ---------------------------------------------------------------------------
__global__ __launch_bounds__(256)
void proj_kernel(const float* __restrict__ X, const float* __restrict__ W,
                 _Float16* __restrict__ Out, int rows, int remap) {
  const int lane = threadIdx.x & 31;
  const int wave = threadIdx.x >> 5;
  const int l16  = lane & 15;
  const int half = lane >> 4;
  const int row0 = (blockIdx.x * 8 + wave) * 16;
  if (row0 >= rows) return;

  // A tile: 16 rows x 64 (two K=32 steps), converted f32->f16 on load.
  v16h a[2];
  const float* xrow = X + (size_t)(row0 + l16) * 64;
#pragma unroll
  for (int ks = 0; ks < 2; ++ks) {
    const int kb = ks * 32 + half * 16;
#pragma unroll
    for (int j = 0; j < 8; ++j) {
      a[ks][j]     = (_Float16)xrow[kb + j];
      a[ks][8 + j] = (_Float16)xrow[kb + 8 + j];
    }
  }

  v8f c[4] = {v8f{}, v8f{}, v8f{}, v8f{}};
#pragma unroll
  for (int ks = 0; ks < 2; ++ks) {     // A constant across the 4 nt WMMAs
#pragma unroll
    for (int nt = 0; nt < 4; ++nt) {
      const int e = nt * 16 + l16;
      v16h b;  // B[d][e] = W[e][d] -> lane reads row e of W, 16 consecutive d
      const float* wrow = W + (size_t)e * 64 + ks * 32 + half * 16;
#pragma unroll
      for (int j = 0; j < 16; ++j) b[j] = (_Float16)wrow[j];
      c[nt] = (nt == 0) ? wmma16(a[ks], b, c[nt]) : wmma16_ra(a[ks], b, c[nt]);
    }
  }

#pragma unroll
  for (int nt = 0; nt < 4; ++nt) {
    const int e = nt * 16 + l16;
#pragma unroll
    for (int r = 0; r < 8; ++r) {
      const int rr = row0 + r + 8 * half;
      size_t orow;
      if (remap) {  // rr = (n*512 + kl)*8 + h  ->  ((n*8+h)*512 + kl)
        const int h  = rr & 7;
        const int kl = (rr >> 3) & 511;
        const int n  = rr >> 12;
        orow = (size_t)((n * 8 + h) * 512 + kl);
      } else {
        orow = (size_t)rr;
      }
      Out[orow * 64 + e] = (_Float16)c[nt][r];
    }
  }
}

// ---------------------------------------------------------------------------
// Kernel 2: flash attention per (n,h).
// Q: [(n*2048+m)*8+h][64] f16,  K/V: [n][h][512][64] f16,
// mask: [n][512] int, O: [n*2048+m][512] f16 (col = h*64+d).
// Block = 256 thr (8 waves), each wave owns a 16-row Q tile; grid (16, 32).
// ---------------------------------------------------------------------------
__global__ __launch_bounds__(256)
void attn_kernel(const _Float16* __restrict__ Qp, const _Float16* __restrict__ Kp,
                 const _Float16* __restrict__ Vp, const int* __restrict__ maskp,
                 _Float16* __restrict__ Op) {
  // All row pitches are multiples of 16B so every operand fetch is b128.
  __shared__ __align__(16) _Float16 ksh[128][72];   // K chunk (kl, d)
  __shared__ __align__(16) _Float16 vt[64][136];    // V chunk transposed (d, kl)
  __shared__ __align__(16) _Float16 pw[8][16][40];  // per-wave P relayout
  __shared__ int msk[128];

  const int lane = threadIdx.x & 31;
  const int wave = threadIdx.x >> 5;
  const int l16  = lane & 15;
  const int half = lane >> 4;
  const int n = blockIdx.y >> 3;
  const int h = blockIdx.y & 7;
  const int m0 = blockIdx.x * 128 + wave * 16;

  // Q A-tile (16 x 64, two K=32 steps), reused for all KL chunks.
  v16h qa[2];
  {
    const _Float16* qrow = Qp + ((size_t)((n * 2048 + m0 + l16) * 8 + h)) * 64;
#pragma unroll
    for (int ks = 0; ks < 2; ++ks) {
      const int kb = ks * 32 + half * 16;
      qa[ks] = ld2x8(qrow + kb, qrow + kb + 8);
    }
  }

  float mrun[8], lrun[8];
#pragma unroll
  for (int r = 0; r < 8; ++r) { mrun[r] = -3.0e38f; lrun[r] = 0.0f; }
  v8f acc[4] = {v8f{}, v8f{}, v8f{}, v8f{}};

  const _Float16* kbase = Kp + (size_t)(n * 8 + h) * 512 * 64;
  const _Float16* vbase = Vp + (size_t)(n * 8 + h) * 512 * 64;
  const float scale = 0.044194173824159216f;  // 1/sqrt(512)

  for (int klb = 0; klb < 512; klb += 128) {
    __syncthreads();  // LDS reuse fence

    // --- K chunk: async DMA straight into LDS (128 x 64 f16, B128 = 8 f16) ---
    for (int t = threadIdx.x; t < 128 * 8; t += 256) {
      const int i  = t >> 3;
      const int d8 = (t & 7) * 8;
      async_copy_b128(lds_off(&ksh[i][d8]),
                      (unsigned)(((klb + i) * 64 + d8) * sizeof(_Float16)),
                      kbase);
    }
    // --- V chunk: b128 global read + transposed b16 scatter into LDS ---
    for (int t = threadIdx.x; t < 128 * 8; t += 256) {
      const int i  = t >> 3;
      const int d8 = (t & 7) * 8;
      const v8h vv = *(const v8h*)(vbase + (size_t)(klb + i) * 64 + d8);
#pragma unroll
      for (int j = 0; j < 8; ++j) vt[d8 + j][i] = vv[j];
    }
    if (threadIdx.x < 128) msk[threadIdx.x] = maskp[n * 512 + klb + threadIdx.x];
    wait_async0();
    __syncthreads();

    for (int sub = 0; sub < 4; ++sub) {
      // --- energy: 16 x 32 tile; A constant across the two nt WMMAs ---
      v8f e[2] = {v8f{}, v8f{}};
#pragma unroll
      for (int ks = 0; ks < 2; ++ks) {
#pragma unroll
        for (int nt = 0; nt < 2; ++nt) {
          const int klloc = sub * 32 + nt * 16 + l16;
          const _Float16* kr = &ksh[klloc][ks * 32 + half * 16];
          const v16h b = ld2x8(kr, kr + 8);  // B[d][kl]: 16 consecutive d
          e[nt] = (nt == 0) ? wmma16(qa[ks], b, e[nt])
                            : wmma16_ra(qa[ks], b, e[nt]);
        }
      }
      const int mv0 = msk[sub * 32 + l16];
      const int mv1 = msk[sub * 32 + 16 + l16];
#pragma unroll
      for (int r = 0; r < 8; ++r) {
        e[0][r] = mv0 ? e[0][r] * scale : -1.0e30f;
        e[1][r] = mv1 ? e[1][r] * scale : -1.0e30f;
      }

      // --- online softmax (per row, reduced across the owning 16-lane half) ---
      float pr0[8], pr1[8];
#pragma unroll
      for (int r = 0; r < 8; ++r) {
        float cm = fmaxf(e[0][r], e[1][r]);
        cm = fmaxf(cm, __shfl_xor(cm, 1));
        cm = fmaxf(cm, __shfl_xor(cm, 2));
        cm = fmaxf(cm, __shfl_xor(cm, 4));
        cm = fmaxf(cm, __shfl_xor(cm, 8));
        const float mnew = fmaxf(mrun[r], cm);
        const float fac  = __expf(mrun[r] - mnew);
        mrun[r] = mnew;
        const float p0 = __expf(e[0][r] - mnew);
        const float p1 = __expf(e[1][r] - mnew);
        float rs = p0 + p1;
        rs += __shfl_xor(rs, 1);
        rs += __shfl_xor(rs, 2);
        rs += __shfl_xor(rs, 4);
        rs += __shfl_xor(rs, 8);
        lrun[r] = lrun[r] * fac + rs;
        pr0[r] = p0; pr1[r] = p1;
#pragma unroll
        for (int dt = 0; dt < 4; ++dt) acc[dt][r] *= fac;
      }

      // --- relayout P (C-format -> A-format) through per-wave LDS ---
#pragma unroll
      for (int r = 0; r < 8; ++r) {
        pw[wave][r + 8 * half][l16]      = (_Float16)pr0[r];
        pw[wave][r + 8 * half][16 + l16] = (_Float16)pr1[r];
      }
      asm volatile("s_wait_dscnt 0" ::: "memory");  // same-wave LDS RAW
      const _Float16* pr = &pw[wave][l16][half * 16];
      const v16h ap = ld2x8(pr, pr + 8);

      // --- O += P(16x32) x V(32x64): A constant across the 4 dt WMMAs ---
#pragma unroll
      for (int dt = 0; dt < 4; ++dt) {
        const _Float16* vr = &vt[dt * 16 + l16][sub * 32 + half * 16];
        const v16h b = ld2x8(vr, vr + 8);  // B[kl][d]: 16 consecutive kl
        acc[dt] = (dt == 0) ? wmma16(ap, b, acc[dt])
                            : wmma16_ra(ap, b, acc[dt]);
      }
    }
  }

  // epilogue: O = acc / l
#pragma unroll
  for (int dt = 0; dt < 4; ++dt) {
#pragma unroll
    for (int r = 0; r < 8; ++r) {
      const float o = acc[dt][r] / lrun[r];
      const size_t row = (size_t)n * 2048 + m0 + r + 8 * half;
      Op[row * 512 + h * 64 + dt * 16 + l16] = (_Float16)o;
    }
  }
}

// ---------------------------------------------------------------------------
// Kernel 3: Y(8192x512 f32) = O(8192x512 f16) @ Wo16^T + bo
// Block = 16-row tile async-staged in LDS; 8 waves cover the 512 columns.
// ---------------------------------------------------------------------------
__global__ __launch_bounds__(256)
void outproj_kernel(const _Float16* __restrict__ O, const _Float16* __restrict__ Wo16,
                    const float* __restrict__ bo, float* __restrict__ Y) {
  __shared__ __align__(16) _Float16 ot[16][520];  // 1040 B rows: 16B-aligned
  const int lane = threadIdx.x & 31;
  const int wave = threadIdx.x >> 5;
  const int l16  = lane & 15;
  const int half = lane >> 4;
  const int row0 = blockIdx.x * 16;

  // O tile via async DMA: 16 x 512 f16, 64 B128 transfers per row.
  for (int t = threadIdx.x; t < 16 * 64; t += 256) {
    const int r   = t >> 6;
    const int cc8 = (t & 63) * 8;
    async_copy_b128(lds_off(&ot[r][cc8]),
                    (unsigned)((((row0 + r) * 512) + cc8) * sizeof(_Float16)),
                    O);
  }
  wait_async0();
  __syncthreads();

  const int colbase = wave * 64;
  v8f c[4] = {v8f{}, v8f{}, v8f{}, v8f{}};
  for (int ks = 0; ks < 16; ++ks) {
    const int kb = ks * 32 + half * 16;
    const _Float16* ar = &ot[l16][kb];
    const v16h a = ld2x8(ar, ar + 8);
#pragma unroll
    for (int nt = 0; nt < 4; ++nt) {
      // B[k][e] = Wo[e][k]; f16 row, 32B-aligned -> single v16h load.
      const v16h b = *(const v16h*)(Wo16 +
          (size_t)(colbase + nt * 16 + l16) * 512 + ks * 32 + half * 16);
      c[nt] = (nt == 0) ? wmma16(a, b, c[nt]) : wmma16_ra(a, b, c[nt]);
    }
  }
#pragma unroll
  for (int nt = 0; nt < 4; ++nt) {
    const int col = colbase + nt * 16 + l16;
    const float bias = bo[col];
#pragma unroll
    for (int r = 0; r < 8; ++r)
      Y[(size_t)(row0 + r + 8 * half) * 512 + col] = c[nt][r] + bias;
  }
}

// ---------------------------------------------------------------------------
extern "C" void kernel_launch(void* const* d_in, const int* in_sizes, int n_in,
                              void* d_out, int out_size, void* d_ws, size_t ws_size,
                              hipStream_t stream) {
  const float* values = (const float*)d_in[0];
  const float* keys   = (const float*)d_in[1];
  const float* query  = (const float*)d_in[2];
  const int*   mask   = (const int*)d_in[3];
  const float* Wv     = (const float*)d_in[4];
  const float* Wk     = (const float*)d_in[5];
  const float* Wq     = (const float*)d_in[6];
  const float* Wo     = (const float*)d_in[7];
  const float* bo     = (const float*)d_in[8];
  float* out = (float*)d_out;

  _Float16* ws   = (_Float16*)d_ws;
  _Float16* qws  = ws;                            // 65536 x 64
  _Float16* kws  = qws + (size_t)65536 * 64;      // [n][h][512][64]
  _Float16* vws  = kws + (size_t)16384 * 64;      // [n][h][512][64]
  _Float16* ows  = vws + (size_t)16384 * 64;      // 8192 x 512
  _Float16* wo16 = ows + (size_t)8192 * 512;      // 512 x 512

  cvtw_kernel<<<256, 256, 0, stream>>>(Wo, wo16);
  proj_kernel<<<128, 256, 0, stream>>>(values, Wv, vws, 16384, 1);
  proj_kernel<<<128, 256, 0, stream>>>(keys,   Wk, kws, 16384, 1);
  proj_kernel<<<512, 256, 0, stream>>>(query,  Wq, qws, 65536, 0);
  attn_kernel<<<dim3(16, 32), 256, 0, stream>>>(qws, kws, vws, mask, ows);
  outproj_kernel<<<512, 256, 0, stream>>>(ows, wo16, bo, out);
}
